// MultiHeadAttention_9096740733081
// MI455X (gfx1250) — compile-verified, matrix operand order checked by reading, not run on hardware
//
#include <hip/hip_runtime.h>
#include <hip/hip_bf16.h>
#include <math.h>

// ---------------------------------------------------------------------------
// MHA forward for B=2, S=2048, D=1024, H=16, DH=64 on gfx1250 (CDNA5, wave32)
// bf16 WMMA (v_wmma_f32_16x16x32_bf16) for all GEMMs + flash-attention core.
// Global->LDS staging uses GLOBAL_LOAD_ASYNC_TO_LDS_B128 (ASYNCcnt) when the
// toolchain exposes the builtin; synchronous v8bf copies otherwise.
// ---------------------------------------------------------------------------

typedef __bf16 bf16;
typedef __attribute__((ext_vector_type(16))) __bf16 v16bf;
typedef __attribute__((ext_vector_type(8)))  __bf16 v8bf;
typedef __attribute__((ext_vector_type(4)))  __bf16 v4bf;
typedef __attribute__((ext_vector_type(8)))  float  v8f;
typedef __attribute__((ext_vector_type(4)))  float  v4f;

#define BB 2
#define SS 2048
#define DD 1024
#define HH 16
#define DHH 64
#define MM (BB*SS)   // 4096 rows in the flattened [B*S, D] activations

// ---- global -> LDS 16-byte copy (async if available) -----------------------
#if __has_builtin(__builtin_amdgcn_global_load_async_to_lds_b128)
#define USE_ASYNC_LDS 1
#endif

#ifdef USE_ASYNC_LDS
// builtin signature (from clang diagnostic): param 1 = global (AS1) pointer to
// int __attribute__((vector_size(16))), param 2 = LDS (AS3) pointer, imm, imm.
typedef int v4i_vs __attribute__((vector_size(16)));
typedef __attribute__((address_space(1))) v4i_vs* g_b128_ptr;
typedef __attribute__((address_space(3))) v4i_vs* l_b128_ptr;
#endif

__device__ __forceinline__ void cp16(bf16* lds_dst, const bf16* gsrc) {
#ifdef USE_ASYNC_LDS
  __builtin_amdgcn_global_load_async_to_lds_b128(
      (g_b128_ptr)(void*)gsrc, (l_b128_ptr)(void*)lds_dst,
      /*imm offset=*/0, /*cpol=*/0);
#else
  *(v8bf*)lds_dst = *(const v8bf*)gsrc;
#endif
}

__device__ __forceinline__ void wait_async_lds() {
#ifdef USE_ASYNC_LDS
  asm volatile("s_wait_asynccnt 0x0" ::: "memory");
#endif
}

// ---- bf16 A/B fragment loader (ISA 7.12.2 16-bit 16x32 layout) -------------
// lane 0..15  : row = lane,     K = {0..7, 16..23}
// lane 16..31 : row = lane-16,  K = {8..15, 24..31}
__device__ __forceinline__ v16bf frag_ld(const bf16* p0, int ld) {
  const int lane = threadIdx.x & 31;
  const int r    = lane & 15;
  const int kb   = (lane >> 4) << 3;        // 0 or 8
  const bf16* p  = p0 + (size_t)r * ld + kb;
  v8bf lo = *(const v8bf*)(p);              // K = kb + 0..7
  v8bf hi = *(const v8bf*)(p + 16);         // K = kb + 16..23
  v16bf f;
#pragma unroll
  for (int i = 0; i < 8; ++i) { f[i] = lo[i]; f[i + 8] = hi[i]; }
  return f;
}

__device__ __forceinline__ v8f wmma_bf16(v16bf a, v16bf b, v8f c) {
  return __builtin_amdgcn_wmma_f32_16x16x32_bf16(
      /*neg_a=*/false, a, /*neg_b=*/false, b,
      /*c_mod=*/(short)0, c, /*reuse_a=*/false, /*reuse_b=*/false);
}

// ---- fp32 -> bf16 conversion (vectorized x4) -------------------------------
__global__ void cvt_f32_to_bf16(const float* __restrict__ in,
                                bf16* __restrict__ out, int n4) {
  int i = blockIdx.x * blockDim.x + threadIdx.x;
  if (i < n4) {
    v4f x = ((const v4f*)in)[i];
    v4bf y;
#pragma unroll
    for (int j = 0; j < 4; ++j) y[j] = (bf16)x[j];
    ((v4bf*)out)[i] = y;
  }
}

// ---- GEMM: Y = A(M,K) * W(N,K)^T + bias ------------------------------------
// MODE 0: write bf16 in per-head layout [B,H,S,DH]   (QKV projections)
// MODE 1: write fp32 flat [M,N]                      (output projection)
// Block: 256 threads = 8 waves (2 along M x 4 along N), block tile 128x128,
// wave tile 64x32 = 4x2 WMMA accumulators, K step 32, A/W tiles staged in LDS.
template <int MODE>
__global__ __launch_bounds__(256) void gemm_bias(
    const bf16* __restrict__ A, const bf16* __restrict__ W,
    const float* __restrict__ bias,
    bf16* __restrict__ outH, float* __restrict__ outF) {
  __shared__ __align__(16) bf16 At[128 * 32];   // [m][k] tile, ld = 32
  __shared__ __align__(16) bf16 Wt[128 * 32];   // [n][k] tile, ld = 32

  const int lane = threadIdx.x & 31;
  const int wave = threadIdx.x >> 5;        // 0..7
  const int wr   = wave >> 2;               // 2 waves along M
  const int wc   = wave & 3;                // 4 waves along N
  const int rowB = blockIdx.y * 128;        // block row base
  const int colB = blockIdx.x * 128;        // block col base
  const int row0 = wr * 64;                 // wave row offset in tile
  const int col0 = wc * 32;                 // wave col offset in tile

  // cooperative-copy mapping: thread -> tile row + 16-element segment
  const int crow = threadIdx.x >> 1;        // 0..127
  const int cseg = (threadIdx.x & 1) * 16;  // element offset 0 or 16

  v8f acc[4][2] = {};

  for (int k0 = 0; k0 < DD; k0 += 32) {
    __syncthreads();                        // previous iteration reads done
    {
      const bf16* gA = A + (size_t)(rowB + crow) * DD + k0 + cseg;
      cp16(&At[crow * 32 + cseg],     gA);
      cp16(&At[crow * 32 + cseg + 8], gA + 8);
      const bf16* gW = W + (size_t)(colB + crow) * DD + k0 + cseg;
      cp16(&Wt[crow * 32 + cseg],     gW);
      cp16(&Wt[crow * 32 + cseg + 8], gW + 8);
      __builtin_prefetch(gA + 32, 0, 3);    // next K-step
      __builtin_prefetch(gW + 32, 0, 3);
    }
    wait_async_lds();
    __syncthreads();

    v16bf a[4], b[2];
#pragma unroll
    for (int i = 0; i < 4; ++i) a[i] = frag_ld(&At[(row0 + 16 * i) * 32], 32);
#pragma unroll
    for (int j = 0; j < 2; ++j) b[j] = frag_ld(&Wt[(col0 + 16 * j) * 32], 32);
#pragma unroll
    for (int i = 0; i < 4; ++i)
#pragma unroll
      for (int j = 0; j < 2; ++j)
        acc[i][j] = wmma_bf16(a[i], b[j], acc[i][j]);
  }

  const int hi = lane >> 4, ln = lane & 15;
#pragma unroll
  for (int ti = 0; ti < 4; ++ti)
#pragma unroll
    for (int tj = 0; tj < 2; ++tj) {
      const int col = colB + col0 + tj * 16 + ln;
      const float bv = bias[col];
#pragma unroll
      for (int r = 0; r < 8; ++r) {
        const int row = rowB + row0 + ti * 16 + r + hi * 8;  // C: M = r(+8)
        const float val = acc[ti][tj][r] + bv;
        if (MODE == 0) {
          const int b_ = row >> 11, s = row & (SS - 1);      // S = 2048
          const int h  = col >> 6,  d = col & (DHH - 1);
          outH[((size_t)(b_ * HH + h) * SS + s) * DHH + d] = (bf16)val;
        } else {
          outF[(size_t)row * DD + col] = val;
        }
      }
    }
}

// ---- Flash attention core --------------------------------------------------
// grid: (S/128, B*H); block 256 = 8 waves, each wave owns 16 q-rows x DH=64.
// Loop over 64-key blocks (causal): scores = Q K^T via WMMA, online softmax,
// O += P V via WMMA with P re-fragmented through LDS.
__global__ __launch_bounds__(256) void flash_attn(
    const bf16* __restrict__ Qh, const bf16* __restrict__ Kh,
    const bf16* __restrict__ Vh, bf16* __restrict__ ctx) {
  __shared__ __align__(16) bf16 Kt[64 * 64];        // [k][d]
  __shared__ __align__(16) bf16 Vt[64 * 64];        // [d][k] (transposed)
  __shared__ __align__(16) bf16 Pb[8][16 * 64];     // per-wave P tile [q][k]

  const int lane = threadIdx.x & 31;
  const int wave = threadIdx.x >> 5;
  const int hi = lane >> 4, ln = lane & 15;
  const int bh = blockIdx.y;                        // b*H + h
  const int q0 = blockIdx.x * 128 + wave * 16;      // this wave's q rows

  const bf16* Qp = Qh + (size_t)bh * SS * DHH;
  const bf16* Kp = Kh + (size_t)bh * SS * DHH;
  const bf16* Vp = Vh + (size_t)bh * SS * DHH;

  // Q fragments for this wave (reused across all key blocks)
  const v16bf qa0 = frag_ld(Qp + (size_t)q0 * DHH, DHH);       // d = 0..31
  const v16bf qa1 = frag_ld(Qp + (size_t)q0 * DHH + 32, DHH);  // d = 32..63

  v8f acc[4] = {};                 // O tiles over d: 4 x (16x16)
  float mrow[8], lrow[8];
#pragma unroll
  for (int r = 0; r < 8; ++r) { mrow[r] = -INFINITY; lrow[r] = 0.f; }

  const float scale = 0.125f;      // 1/sqrt(DH)
  const int nkb = (blockIdx.x + 1) * 2;   // causal: keys up to q-tile end

  for (int kb = 0; kb < nkb; ++kb) {
    const int kbase = kb * 64;
    __syncthreads();               // previous iteration's LDS reads done
    {
      // thread t -> KV row t/4, 16 d-columns at (t%4)*16
      const int krow = threadIdx.x >> 2;
      const int c0   = (threadIdx.x & 3) * 16;
      const bf16* gK = Kp + (size_t)(kbase + krow) * DHH + c0;
      cp16(&Kt[krow * 64 + c0],     gK);        // async K tile
      cp16(&Kt[krow * 64 + c0 + 8], gK + 8);
      const v8bf* vs = (const v8bf*)(Vp + (size_t)(kbase + krow) * DHH + c0);
      v8bf y0 = vs[0], y1 = vs[1];
#pragma unroll
      for (int i = 0; i < 8; ++i) {             // scatter V transposed
        Vt[(c0 + i) * 64 + krow]     = y0[i];
        Vt[(c0 + 8 + i) * 64 + krow] = y1[i];
      }
    }
    wait_async_lds();
    __syncthreads();

    // ---- scores S = Q K^T (4 tiles of 16 keys, contraction over d=64) ----
    v8f sc[4];
#pragma unroll
    for (int n = 0; n < 4; ++n) {
      v16bf kf0 = frag_ld(&Kt[n * 16 * 64], 64);        // d = 0..31
      v16bf kf1 = frag_ld(&Kt[n * 16 * 64 + 32], 64);   // d = 32..63
      v8f z = {};
      z = wmma_bf16(qa0, kf0, z);
      z = wmma_bf16(qa1, kf1, z);
      sc[n] = z;
    }

    // ---- scale + causal mask --------------------------------------------
#pragma unroll
    for (int n = 0; n < 4; ++n) {
      const int col = kbase + n * 16 + ln;
#pragma unroll
      for (int r = 0; r < 8; ++r) {
        const int rowg = q0 + r + hi * 8;
        const float vS = sc[n][r] * scale;
        sc[n][r] = (col <= rowg) ? vS : -INFINITY;
      }
    }

    // ---- online softmax (row stats replicated across each lane half) ----
#pragma unroll
    for (int r = 0; r < 8; ++r) {
      float rm = fmaxf(fmaxf(sc[0][r], sc[1][r]), fmaxf(sc[2][r], sc[3][r]));
#pragma unroll
      for (int off = 1; off < 16; off <<= 1)
        rm = fmaxf(rm, __shfl_xor(rm, off, 32));        // within half-wave
      const float mn   = fmaxf(mrow[r], rm);
      const float safe = (mn == -INFINITY) ? 0.f : mn;  // fully-masked guard
      const float corr = __expf(mrow[r] - safe);
      float rs = 0.f;
#pragma unroll
      for (int n = 0; n < 4; ++n) {
        const float p = __expf(sc[n][r] - safe);
        sc[n][r] = p;
        rs += p;
      }
#pragma unroll
      for (int off = 1; off < 16; off <<= 1)
        rs += __shfl_xor(rs, off, 32);
      lrow[r] = lrow[r] * corr + rs;
      mrow[r] = mn;
#pragma unroll
      for (int n = 0; n < 4; ++n) acc[n][r] *= corr;
    }

    // ---- stage P (C layout -> LDS row-major) then re-fragment as A ------
#pragma unroll
    for (int n = 0; n < 4; ++n)
#pragma unroll
      for (int r = 0; r < 8; ++r)
        Pb[wave][(r + hi * 8) * 64 + n * 16 + ln] = (bf16)sc[n][r];
    __syncthreads();

    const v16bf pf0 = frag_ld(&Pb[wave][0],  64);       // k = 0..31
    const v16bf pf1 = frag_ld(&Pb[wave][32], 64);       // k = 32..63
#pragma unroll
    for (int n = 0; n < 4; ++n) {
      v16bf vf0 = frag_ld(&Vt[n * 16 * 64], 64);        // B: d cols, k 0..31
      v16bf vf1 = frag_ld(&Vt[n * 16 * 64 + 32], 64);   // k 32..63
      acc[n] = wmma_bf16(pf0, vf0, acc[n]);
      acc[n] = wmma_bf16(pf1, vf1, acc[n]);
    }
  }

  // ---- normalize and write context in [B,S,D] layout (bf16) --------------
  const int b = bh / HH, h = bh % HH;
#pragma unroll
  for (int r = 0; r < 8; ++r) {
    const float inv = (lrow[r] > 0.f) ? (1.f / lrow[r]) : 0.f;
    const int s = q0 + r + hi * 8;
    bf16* dst = ctx + (size_t)(b * SS + s) * DD + h * DHH;
#pragma unroll
    for (int n = 0; n < 4; ++n)
      dst[n * 16 + ln] = (bf16)(acc[n][r] * inv);
  }
}

// ---------------------------------------------------------------------------
extern "C" void kernel_launch(void* const* d_in, const int* in_sizes, int n_in,
                              void* d_out, int out_size, void* d_ws,
                              size_t ws_size, hipStream_t stream) {
  const float* q  = (const float*)d_in[0];
  const float* k  = (const float*)d_in[1];
  const float* v  = (const float*)d_in[2];
  // d_in[3] = mask (bool tril) -- causality is computed analytically
  const float* Wq = (const float*)d_in[4];
  const float* bq = (const float*)d_in[5];
  const float* Wk = (const float*)d_in[6];
  const float* bk = (const float*)d_in[7];
  const float* Wv = (const float*)d_in[8];
  const float* bv = (const float*)d_in[9];
  const float* Wo = (const float*)d_in[10];
  const float* bo = (const float*)d_in[11];
  float* out = (float*)d_out;

  // workspace carve (all bf16): 7*XD + 4*WD = 32M elems = 64 MB
  bf16* ws = (bf16*)d_ws;
  const size_t XD = (size_t)MM * DD;        // 4096*1024
  const size_t WD = (size_t)DD * DD;        // 1024*1024
  bf16* qb  = ws;        bf16* kb  = qb  + XD; bf16* vb  = kb  + XD;
  bf16* wqb = vb + XD;   bf16* wkb = wqb + WD; bf16* wvb = wkb + WD;
  bf16* wob = wvb + WD;
  bf16* Qhp = wob + WD;  bf16* Khp = Qhp + XD; bf16* Vhp = Khp + XD;
  bf16* ctx = Vhp + XD;

  const int thr = 256;
  auto cvt = [&](const float* src, bf16* dst, size_t n) {
    const int n4 = (int)(n / 4);
    hipLaunchKernelGGL(cvt_f32_to_bf16, dim3((n4 + thr - 1) / thr), dim3(thr),
                       0, stream, src, dst, n4);
  };
  cvt(q, qb, XD);  cvt(k, kb, XD);  cvt(v, vb, XD);
  cvt(Wq, wqb, WD); cvt(Wk, wkb, WD); cvt(Wv, wvb, WD); cvt(Wo, wob, WD);

  const dim3 ggrid(DD / 128, MM / 128);     // 8 x 32 blocks
  hipLaunchKernelGGL((gemm_bias<0>), ggrid, dim3(256), 0, stream,
                     qb, wqb, bq, Qhp, (float*)nullptr);
  hipLaunchKernelGGL((gemm_bias<0>), ggrid, dim3(256), 0, stream,
                     kb, wkb, bk, Khp, (float*)nullptr);
  hipLaunchKernelGGL((gemm_bias<0>), ggrid, dim3(256), 0, stream,
                     vb, wvb, bv, Vhp, (float*)nullptr);

  hipLaunchKernelGGL(flash_attn, dim3(SS / 128, BB * HH), dim3(256), 0, stream,
                     Qhp, Khp, Vhp, ctx);

  hipLaunchKernelGGL((gemm_bias<1>), ggrid, dim3(256), 0, stream,
                     ctx, wob, bo, (bf16*)nullptr, out);
}
